// Encoder_35442070126950
// MI455X (gfx1250) — compile-verified, matrix operand order checked by reading, not run on hardware
//
#include <hip/hip_runtime.h>
#include <hip/hip_bf16.h>
#include <math.h>

// ---------------------------------------------------------------------------
// Types for CDNA5 WMMA (wave32): D(f32 16x16) = A(bf16 16x32) x B(bf16 32x16) + C
// ---------------------------------------------------------------------------
typedef __bf16 bf16_t;
typedef __attribute__((ext_vector_type(16))) __bf16 v16bf;
typedef __attribute__((ext_vector_type(8)))  __bf16 v8bf;
typedef __attribute__((ext_vector_type(8)))  float  v8f;

#define WAVES_PER_BLOCK 8
#define TBLOCK (32 * WAVES_PER_BLOCK)
#define MT 4                                  // M-tiles (of 16 rows) per wave

static __device__ __forceinline__ v8bf zero8() {
    v8bf z;
#pragma unroll
    for (int i = 0; i < 8; ++i) z[i] = (__bf16)0.f;
    return z;
}

// ---------------------------------------------------------------------------
// fp32 -> bf16 convert (grid-stride)
// ---------------------------------------------------------------------------
__global__ void cvt_f32_bf16(const float* __restrict__ s, bf16_t* __restrict__ d, long n) {
    long i  = (long)blockIdx.x * blockDim.x + threadIdx.x;
    long st = (long)gridDim.x * blockDim.x;
    for (; i < n; i += st) d[i] = (bf16_t)s[i];
}

// Weights [K][Cout] (HWIO flat) -> transposed bf16 [CoutPad][K], zero-filled pad rows.
__global__ void cvt_transpose_w(const float* __restrict__ w, bf16_t* __restrict__ wT,
                                int K, int Cout, int CoutPad) {
    long n  = (long)K * CoutPad;
    long i  = (long)blockIdx.x * blockDim.x + threadIdx.x;
    long st = (long)gridDim.x * blockDim.x;
    for (; i < n; i += st) {
        int co = (int)(i / K);
        int kg = (int)(i - (long)co * K);
        wT[i] = (co < Cout) ? (bf16_t)w[(long)kg * Cout + co] : (bf16_t)0.f;
    }
}

// ---------------------------------------------------------------------------
// Fast implicit-GEMM conv for layers with Cin % 32 == 0 (all but the head).
// MODE 0: zero pad, stride S.  MODE 1: reflect pad, stride 1.  MODE 2: transposed s2.
// Wave tile: 64 (M = 4 x 16 flat b*oy*ox) x 16 (N = cout).  K chunked by 32; each
// chunk lies inside one (kh,kw) tap, so a per-lane A fragment = two contiguous
// b128 loads.  The 32K x 16N weight slab is double-buffered in LDS (transposed
// [n][k], 40-elt rows => 16B-aligned, conflict-free) and staged by wave 0 with
// global_load_async_to_lds_b128 (ASYNCcnt), overlapping the next slab's DMA
// with the current 4-WMMA body.  One workgroup barrier per chunk.
// ---------------------------------------------------------------------------
template <int MODE>
__global__ void __launch_bounds__(TBLOCK)
conv_wmma_fast(const bf16_t* __restrict__ in, const bf16_t* __restrict__ wT,
               const float* __restrict__ bias, float* __restrict__ out,
               int Bn, int Hin, int Win, int Cin, int Ho, int Wo, int Cout,
               int K, int KW, int stride, int pad)
{
    __shared__ bf16_t smBT[2][16][40];       // double-buffered [n][k]

    const int M    = Bn * Ho * Wo;
    const int tid  = threadIdx.x;
    const int lane = tid & 31;
    const int wave = tid >> 5;

    const int mTile0 = (blockIdx.x * WAVES_PER_BLOCK + wave) * MT;
    const int n0     = blockIdx.y * 16;

    const int mRow  = lane & 15;
    const int kBase = (lane < 16) ? 0 : 8;   // K-half owned by this lane group

    int  bI[MT], oyS[MT], oxS[MT];
    bool mOK[MT];
#pragma unroll
    for (int t = 0; t < MT; ++t) {
        int gm = (mTile0 + t) * 16 + mRow;
        mOK[t] = (gm < M);
        int b = 0, oy = 0, ox = 0;
        if (mOK[t]) {
            b = gm / (Ho * Wo);
            int r = gm - b * Ho * Wo;
            oy = r / Wo;
            ox = r - oy * Wo;
        }
        bI[t]  = b;
        oyS[t] = (MODE == 0) ? oy * stride - pad : oy - pad;
        oxS[t] = (MODE == 0) ? ox * stride - pad : ox - pad;
    }

    const int nCol = n0 + (lane & 15);
    v8f acc[MT];
    {
        float bv = (nCol < Cout) ? bias[nCol] : 0.f;
#pragma unroll
        for (int t = 0; t < MT; ++t)
#pragma unroll
            for (int r = 0; r < 8; ++r) acc[t][r] = bv;
    }

    const int nChunks = K >> 5;              // K % 32 == 0 here

    auto stage = [&](int kc) {               // wave 0 only: async 1KB slab -> LDS
        const long kg0 = (long)kc << 5;
        const int  buf = kc & 1;
#pragma unroll
        for (int t = lane; t < 64; t += 32) {        // 64 x 16B
            int n  = t >> 2;
            int kq = (t & 3) << 3;                   // 0,8,16,24
            unsigned ldsoff = (unsigned)(unsigned long long)&smBT[buf][n][kq];
            unsigned long long gaddr =
                (unsigned long long)(wT + (long)(n0 + n) * K + kg0 + kq);
            asm volatile("global_load_async_to_lds_b128 %0, %1, off"
                         :: "v"(ldsoff), "v"(gaddr) : "memory");
        }
    };

    if (wave == 0) stage(0);

    int kh = 0, kw = 0, ciBase = 0;
    for (int kc = 0; kc < nChunks; ++kc) {
        if (wave == 0) asm volatile("s_wait_asynccnt 0x0" ::: "memory");
        __syncthreads();      // publish buf[kc&1]; all prior reads of other buf done
        if (wave == 0 && kc + 1 < nChunks) stage(kc + 1);   // overlap next DMA

        const int buf = kc & 1;
        // ---- B fragment: two ds_load_b128 from the transposed LDS slab ----
        const v8bf bLo = *reinterpret_cast<const v8bf*>(&smBT[buf][lane & 15][kBase]);
        const v8bf bHi = *reinterpret_cast<const v8bf*>(&smBT[buf][lane & 15][kBase + 16]);
        v16bf bf;
#pragma unroll
        for (int e = 0; e < 8; ++e) { bf[e] = bLo[e]; bf[8 + e] = bHi[e]; }

        // ---- 4 M-tiles: A fragment = two contiguous b128 loads each ----
#pragma unroll
        for (int t = 0; t < MT; ++t) {
            int iy, ix; bool ok = mOK[t];
            if (MODE == 2) {                 // transposed conv, stride 2
                int ty = oyS[t] + kh, tx = oxS[t] + kw;
                ok = ok && (ty >= 0) && (tx >= 0) && !(ty & 1) && !(tx & 1);
                iy = ty >> 1; ix = tx >> 1;
                ok = ok && (iy < Hin) && (ix < Win);
            } else if (MODE == 1) {          // reflect pad
                iy = oyS[t] + kh; ix = oxS[t] + kw;
                iy = (iy < 0) ? -iy : iy;  iy = (iy >= Hin) ? (2 * Hin - 2 - iy) : iy;
                ix = (ix < 0) ? -ix : ix;  ix = (ix >= Win) ? (2 * Win - 2 - ix) : ix;
            } else {                         // zero pad
                iy = oyS[t] + kh; ix = oxS[t] + kw;
                ok = ok && (iy >= 0) && (iy < Hin) && (ix >= 0) && (ix < Win);
            }
            v8bf aLo = zero8(), aHi = zero8();
            if (ok) {
                const bf16_t* p = in + (((long)bI[t] * Hin + iy) * Win + ix) * Cin
                                     + ciBase + kBase;
                aLo = *reinterpret_cast<const v8bf*>(p);
                aHi = *reinterpret_cast<const v8bf*>(p + 16);
            }
            v16bf a;
#pragma unroll
            for (int e = 0; e < 8; ++e) { a[e] = aLo[e]; a[8 + e] = aHi[e]; }
            acc[t] = __builtin_amdgcn_wmma_f32_16x16x32_bf16(
                         false, a, false, bf, (short)0, acc[t], false, false);
        }

        // advance tap / ci window (no divides in the loop)
        ciBase += 32;
        if (ciBase >= Cin) { ciBase = 0; if (++kw == KW) { kw = 0; ++kh; } }
    }

    // D layout: VGPR r -> M = mBase + r, N = lane&15 (lanes 16-31: mBase = 8)
    const int mBase = (lane < 16) ? 0 : 8;
    if (nCol < Cout) {
#pragma unroll
        for (int t = 0; t < MT; ++t)
#pragma unroll
            for (int r = 0; r < 8; ++r) {
                int g = (mTile0 + t) * 16 + mBase + r;
                if (g < M) out[(long)g * Cout + nCol] = acc[t][r];
            }
    }
}

// ---------------------------------------------------------------------------
// Gather-style conv for the head layer (Cin = 3, reflect pad).  CIN/KWc are
// template constants so the k -> (kh,kw,ci) decomposition strength-reduces to
// mul-shift (no runtime integer division in the unrolled body).
// ---------------------------------------------------------------------------
template <int CIN, int KWc>
__global__ void __launch_bounds__(TBLOCK)
conv_wmma_gather(const bf16_t* __restrict__ in, const bf16_t* __restrict__ wT,
                 const float* __restrict__ bias, float* __restrict__ out,
                 int Bn, int Hin, int Win, int Ho, int Wo, int Cout,
                 int K, int pad)
{
    const int M    = Bn * Ho * Wo;
    const int tid  = threadIdx.x;
    const int lane = tid & 31;
    const int wave = tid >> 5;

    const int mTile = blockIdx.x * WAVES_PER_BLOCK + wave;
    const int n0    = blockIdx.y * 16;

    const int mRow  = lane & 15;
    const int kBase = (lane < 16) ? 0 : 8;
    const int gm    = mTile * 16 + mRow;
    const bool mOK  = (gm < M);

    int b = 0, oy = 0, ox = 0;
    if (mOK) {
        b = gm / (Ho * Wo);
        int r = gm - b * Ho * Wo;
        oy = r / Wo;
        ox = r - oy * Wo;
    }

    const int nCol = n0 + (lane & 15);
    const long wRow = (long)nCol * K;
    v8f acc;
    {
        float bv = (nCol < Cout) ? bias[nCol] : 0.f;
#pragma unroll
        for (int r = 0; r < 8; ++r) acc[r] = bv;
    }

    const int nChunks = (K + 31) >> 5;
    for (int kc = 0; kc < nChunks; ++kc) {
        if (kc + 1 < nChunks)
            __builtin_prefetch(&wT[wRow + ((kc + 1) << 5)], 0, 1);
        v16bf a, bf;
#pragma unroll
        for (int i = 0; i < 8; ++i) {
#pragma unroll
            for (int j = 0; j < 2; ++j) {
                const int khalf = (i < 4) ? 2 * i : 16 + 2 * (i - 4);
                const int k  = khalf + kBase + j;
                const int e  = i * 2 + j;
                const int kg = (kc << 5) + k;
                bf16_t av = (bf16_t)0.f, bv = (bf16_t)0.f;
                if (kg < K) {
                    bv = wT[wRow + kg];              // pad rows zero-filled
                    if (mOK) {
                        int kh  = kg / (KWc * CIN);  // constant divisors
                        int rem = kg - kh * (KWc * CIN);
                        int kw  = rem / CIN;
                        int ci  = rem - kw * CIN;
                        int iy = oy + kh - pad, ix = ox + kw - pad;
                        iy = (iy < 0) ? -iy : iy;  iy = (iy >= Hin) ? (2 * Hin - 2 - iy) : iy;
                        ix = (ix < 0) ? -ix : ix;  ix = (ix >= Win) ? (2 * Win - 2 - ix) : ix;
                        av = in[(((long)b * Hin + iy) * Win + ix) * CIN + ci];
                    }
                }
                a[e] = av; bf[e] = bv;
            }
        }
        acc = __builtin_amdgcn_wmma_f32_16x16x32_bf16(
                  false, a, false, bf, (short)0, acc, false, false);
    }

    const int mBase = (lane < 16) ? 0 : 8;
    if (nCol < Cout) {
#pragma unroll
        for (int r = 0; r < 8; ++r) {
            int g = mTile * 16 + mBase + r;
            if (g < M) out[(long)g * Cout + nCol] = acc[r];
        }
    }
}

// ---------------------------------------------------------------------------
// BatchNorm reductions: per-channel sum/sumsq via LDS float atomics, then one
// global atomic per channel per block.
// ---------------------------------------------------------------------------
__global__ void bn_reduce(const float* __restrict__ x, float* __restrict__ sums,
                          float* __restrict__ sqs, long n, int C)
{
    __shared__ float sS[512];
    __shared__ float sQ[512];
    for (int c = threadIdx.x; c < C; c += blockDim.x) { sS[c] = 0.f; sQ[c] = 0.f; }
    __syncthreads();
    long i  = (long)blockIdx.x * blockDim.x + threadIdx.x;
    long st = (long)gridDim.x * blockDim.x;
    for (; i < n; i += st) {
        float v = x[i];
        int c = (int)(i % C);
        atomicAdd(&sS[c], v);          // ds_add_f32
        atomicAdd(&sQ[c], v * v);
    }
    __syncthreads();
    for (int c = threadIdx.x; c < C; c += blockDim.x) {
        atomicAdd(&sums[c], sS[c]);
        atomicAdd(&sqs[c],  sQ[c]);
    }
}

__global__ void bn_apply_relu(const float* __restrict__ x, bf16_t* __restrict__ y,
                              const float* __restrict__ sums, const float* __restrict__ sqs,
                              const float* __restrict__ g, const float* __restrict__ bta,
                              long n, int C, float invCount)
{
    long i  = (long)blockIdx.x * blockDim.x + threadIdx.x;
    long st = (long)gridDim.x * blockDim.x;
    for (; i < n; i += st) {
        int c = (int)(i % C);
        float m = sums[c] * invCount;
        float v = sqs[c] * invCount - m * m;
        float t = g[c] * (x[i] - m) * rsqrtf(v + 1e-3f) + bta[c];
        y[i] = (bf16_t)fmaxf(t, 0.f);
    }
}

// ---------------------------------------------------------------------------
// Instance-wise mean: fused tanh + segment accumulate, then broadcast.
// ---------------------------------------------------------------------------
__global__ void inst_accum(const float* __restrict__ x, const int* __restrict__ inst,
                           float* __restrict__ isum, float* __restrict__ icnt,
                           int Bn, int HW, int NI, int C)
{
    long n  = (long)Bn * HW;
    long i  = (long)blockIdx.x * blockDim.x + threadIdx.x;
    long st = (long)gridDim.x * blockDim.x;
    for (; i < n; i += st) {
        int b   = (int)(i / HW);
        int seg = b * NI + inst[i];
        atomicAdd(&icnt[seg], 1.f);
        for (int c = 0; c < C; ++c)
            atomicAdd(&isum[seg * C + c], tanhf(x[i * C + c]));
    }
}

__global__ void inst_bcast(const int* __restrict__ inst, const float* __restrict__ isum,
                           const float* __restrict__ icnt, float* __restrict__ out,
                           int Bn, int HW, int NI, int C)
{
    long n  = (long)Bn * HW;
    long i  = (long)blockIdx.x * blockDim.x + threadIdx.x;
    long st = (long)gridDim.x * blockDim.x;
    for (; i < n; i += st) {
        int b   = (int)(i / HW);
        int seg = b * NI + inst[i];
        float ic = fmaxf(icnt[seg], 1.f);
        for (int c = 0; c < C; ++c)
            out[i * C + c] = isum[seg * C + c] / ic;
    }
}

// ---------------------------------------------------------------------------
// Host orchestration
// ---------------------------------------------------------------------------
static inline unsigned gsz(long n, int tb, long cap) {
    long b = (n + tb - 1) / tb;
    if (b < 1) b = 1;
    if (b > cap) b = cap;
    return (unsigned)b;
}

extern "C" void kernel_launch(void* const* d_in, const int* in_sizes, int n_in,
                              void* d_out, int out_size, void* d_ws, size_t ws_size,
                              hipStream_t stream)
{
    (void)in_sizes; (void)n_in; (void)out_size; (void)ws_size;
    const int B = 4, S = 512, NI = 32;

    const float* x_f32 = (const float*)d_in[0];
    const int*   inst  = (const int*)d_in[1];

    const float* conv_w[10]; const float* conv_b[10];
    const float* bn_g[9];    const float* bn_b[9];
    int pi = 2;
    conv_w[0] = (const float*)d_in[pi++]; conv_b[0] = (const float*)d_in[pi++];
    bn_g[0]   = (const float*)d_in[pi++]; bn_b[0]   = (const float*)d_in[pi++];
    for (int i = 0; i < 4; ++i) {
        conv_w[1 + i] = (const float*)d_in[pi++]; conv_b[1 + i] = (const float*)d_in[pi++];
        bn_g[1 + i]   = (const float*)d_in[pi++]; bn_b[1 + i]   = (const float*)d_in[pi++];
    }
    for (int i = 0; i < 4; ++i) {
        conv_w[5 + i] = (const float*)d_in[pi++]; conv_b[5 + i] = (const float*)d_in[pi++];
        bn_g[5 + i]   = (const float*)d_in[pi++]; bn_b[5 + i]   = (const float*)d_in[pi++];
    }
    conv_w[9] = (const float*)d_in[pi++]; conv_b[9] = (const float*)d_in[pi++];

    struct L { int Hin, Cin, Ho, Cout, KH, stride, pad, mode; };
    const L ls[10] = {
        {512,   3, 512,  32, 7, 1, 3, 1},   // head, reflect pad (gather path)
        {512,  32, 256,  64, 3, 2, 0, 0},   // down (SAME, pad_lo=0)
        {256,  64, 128, 128, 3, 2, 0, 0},
        {128, 128,  64, 256, 3, 2, 0, 0},
        { 64, 256,  32, 512, 3, 2, 0, 0},
        { 32, 512,  64, 256, 3, 1, 1, 2},   // transposed, stride 2
        { 64, 256, 128, 128, 3, 1, 1, 2},
        {128, 128, 256,  64, 3, 1, 1, 2},
        {256,  64, 512,  32, 3, 1, 1, 2},
        {512,  32, 512,   3, 7, 1, 3, 1},   // tail, reflect pad (fast path)
    };

    // -------- workspace layout --------
    char* ws = (char*)d_ws;
    size_t off = 0;
    auto alloc = [&](size_t bytes) { size_t o = off; off += (bytes + 255) & ~(size_t)255; return o; };

    const size_t xbf_off = alloc((size_t)B * S * S * 3 * sizeof(bf16_t));
    size_t wT_off[10]; int Ks[10], CoutPad[10];
    for (int l = 0; l < 10; ++l) {
        Ks[l]      = ls[l].KH * ls[l].KH * ls[l].Cin;
        CoutPad[l] = ((ls[l].Cout + 15) / 16) * 16;
        wT_off[l]  = alloc((size_t)Ks[l] * CoutPad[l] * sizeof(bf16_t));
    }
    const size_t maxAct   = (size_t)B * S * S * 32;
    const size_t actA_off = alloc(maxAct * sizeof(float));
    const size_t actB_off = alloc(maxAct * sizeof(bf16_t));
    const size_t red_off  = alloc(2 * 512 * sizeof(float));
    const size_t isum_off = alloc((size_t)B * NI * 3 * sizeof(float));
    const size_t icnt_off = alloc((size_t)B * NI * sizeof(float));

    // -------- converts: input -> bf16, weights -> transposed bf16 --------
    {
        long nx = (long)B * S * S * 3;
        cvt_f32_bf16<<<gsz(nx, 256, 4096), 256, 0, stream>>>(x_f32, (bf16_t*)(ws + xbf_off), nx);
        for (int l = 0; l < 10; ++l) {
            long we = (long)Ks[l] * CoutPad[l];
            cvt_transpose_w<<<gsz(we, 256, 2048), 256, 0, stream>>>(
                conv_w[l], (bf16_t*)(ws + wT_off[l]), Ks[l], ls[l].Cout, CoutPad[l]);
        }
    }

    // -------- conv / BN pipeline --------
    const bf16_t* cur = (const bf16_t*)(ws + xbf_off);
    float*  actA = (float*)(ws + actA_off);
    bf16_t* actB = (bf16_t*)(ws + actB_off);
    float*  sums = (float*)(ws + red_off);
    float*  sqs  = sums + 512;

    for (int l = 0; l < 10; ++l) {
        const L& q = ls[l];
        const int Hin = q.Hin, Win = q.Hin, Cin = q.Cin;
        const int Ho = q.Ho, Wo = q.Ho, Cout = q.Cout;
        const long M = (long)B * Ho * Wo;
        const bf16_t* wT = (const bf16_t*)(ws + wT_off[l]);
        const unsigned gy = (unsigned)((Cout + 15) / 16);

        if (l == 0) {
            dim3 grid((unsigned)((M / 16 + WAVES_PER_BLOCK - 1) / WAVES_PER_BLOCK), gy);
            conv_wmma_gather<3, 7><<<grid, TBLOCK, 0, stream>>>(
                cur, wT, conv_b[l], actA,
                B, Hin, Win, Ho, Wo, Cout, Ks[l], q.pad);
        } else {
            const long mt = M / 16;
            dim3 grid((unsigned)((mt + (long)MT * WAVES_PER_BLOCK - 1) /
                                 ((long)MT * WAVES_PER_BLOCK)), gy);
            if (q.mode == 0) {
                conv_wmma_fast<0><<<grid, TBLOCK, 0, stream>>>(
                    cur, wT, conv_b[l], actA,
                    B, Hin, Win, Cin, Ho, Wo, Cout, Ks[l], q.KH, q.stride, q.pad);
            } else if (q.mode == 1) {
                conv_wmma_fast<1><<<grid, TBLOCK, 0, stream>>>(
                    cur, wT, conv_b[l], actA,
                    B, Hin, Win, Cin, Ho, Wo, Cout, Ks[l], q.KH, q.stride, q.pad);
            } else {
                conv_wmma_fast<2><<<grid, TBLOCK, 0, stream>>>(
                    cur, wT, conv_b[l], actA,
                    B, Hin, Win, Cin, Ho, Wo, Cout, Ks[l], q.KH, q.stride, q.pad);
            }
        }

        if (l < 9) {
            long n = M * Cout;
            hipMemsetAsync(ws + red_off, 0, 2 * 512 * sizeof(float), stream);
            bn_reduce<<<gsz(n, 256, 2048), 256, 0, stream>>>(actA, sums, sqs, n, Cout);
            bn_apply_relu<<<gsz(n, 256, 4096), 256, 0, stream>>>(
                actA, actB, sums, sqs, bn_g[l], bn_b[l], n, Cout, 1.f / (float)M);
            cur = actB;
        }
    }

    // -------- tanh + instance-wise mean --------
    {
        hipMemsetAsync(ws + isum_off, 0, (size_t)B * NI * 3 * sizeof(float), stream);
        hipMemsetAsync(ws + icnt_off, 0, (size_t)B * NI * sizeof(float), stream);
        float* isum = (float*)(ws + isum_off);
        float* icnt = (float*)(ws + icnt_off);
        long n = (long)B * S * S;
        inst_accum<<<gsz(n, 256, 4096), 256, 0, stream>>>(actA, inst, isum, icnt, B, S * S, NI, 3);
        inst_bcast<<<gsz(n, 256, 4096), 256, 0, stream>>>(inst, isum, icnt, (float*)d_out, B, S * S, NI, 3);
    }
}
